// GPT_82282983457350
// MI455X (gfx1250) — matrix operand order repaired and checked by reading, not compiled, not run on hardware
//
#include <hip/hip_runtime.h>
#include <hip/hip_bf16.h>
#include <math.h>

typedef __attribute__((ext_vector_type(16))) __bf16 v16bf;
typedef __attribute__((ext_vector_type(8)))  float  v8f;

#define B_   2
#define T_   1024
#define D_   768
#define L_   12
#define H_   12
#define DK_  64
#define DFF_ 4096
#define V_   50257

// native f32 -> bf16 (RTNE via hardware cvt)
__device__ __forceinline__ __bf16 f2bf(float f) { return (__bf16)f; }

// 16-bit A-matrix 16x32 per-lane K index pattern (contiguous, matches B fills):
// lanes 0-15 hold K = 0..15 ; lanes 16-31 hold K = 16..31
__device__ __forceinline__ int apat(int hlf, int e) {
  return hlf * 16 + e;
}

// ---- CDNA5 async global->LDS copy (ASYNCcnt) ----
__device__ __forceinline__ void async_b128(unsigned lds_off, const void* gptr) {
  asm volatile("global_load_async_to_lds_b128 %0, %1, off"
               :: "v"(lds_off), "v"((unsigned long long)(size_t)gptr)
               : "memory");
}
__device__ __forceinline__ void wait_async0() {
  asm volatile("s_wait_asynccnt 0x0" ::: "memory");
}
__device__ __forceinline__ unsigned lds_addr(const void* p) {
  return (unsigned)(size_t)p;   // LDS aperture: addr[31:0] is the LDS offset
}

// ---- DPP 16-lane butterfly reductions (rows of the WMMA C layout) ----
template <int CTRL>
__device__ __forceinline__ float dppf(float v) {
  int xi = __builtin_bit_cast(int, v);
  int yi = __builtin_amdgcn_update_dpp(xi, xi, CTRL, 0xF, 0xF, true);
  return __builtin_bit_cast(float, yi);
}
__device__ __forceinline__ float red16_max(float x) {
  x = fmaxf(x, dppf<0xB1>(x));    // quad_perm(1,0,3,2)  : xor 1
  x = fmaxf(x, dppf<0x4E>(x));    // quad_perm(2,3,0,1)  : xor 2
  x = fmaxf(x, dppf<0x141>(x));   // row_half_mirror     : xor 7
  x = fmaxf(x, dppf<0x140>(x));   // row_mirror          : xor 15
  return x;
}
__device__ __forceinline__ float red16_sum(float x) {
  x += dppf<0xB1>(x);
  x += dppf<0x4E>(x);
  x += dppf<0x141>(x);
  x += dppf<0x140>(x);
  return x;
}

// ---------------- embedding: x = tok_emb[idx] + pos_emb ----------------
__global__ void embed_k(const int* __restrict__ idx, const float* __restrict__ tok,
                        const float* __restrict__ pos, float* __restrict__ x) {
  int row = blockIdx.x;            // b*T + t
  int t = row % T_;
  int token = idx[row];
  const float* te = tok + (size_t)token * D_;
  const float* pe = pos + (size_t)t * D_;
  float* xr = x + (size_t)row * D_;
  for (int j = threadIdx.x; j < D_; j += blockDim.x) xr[j] = te[j] + pe[j];
}

// ---------------- LayerNorm (fp32 in -> bf16 out) ----------------
__global__ __launch_bounds__(256)
void ln_bf16(const float* __restrict__ x, const float* __restrict__ w,
             const float* __restrict__ b, __bf16* __restrict__ out) {
  __shared__ float s1[256], s2[256];
  const int row = blockIdx.x, tid = threadIdx.x;
  const float* xr = x + (size_t)row * D_;
  float a = 0.f, q = 0.f;
  for (int j = tid; j < D_; j += 256) { float v = xr[j]; a += v; q += v * v; }
  s1[tid] = a; s2[tid] = q;
  __syncthreads();
  for (int st = 128; st > 0; st >>= 1) {
    if (tid < st) { s1[tid] += s1[tid + st]; s2[tid] += s2[tid + st]; }
    __syncthreads();
  }
  float mu  = s1[0] * (1.0f / D_);
  float var = s2[0] * (1.0f / D_) - mu * mu;
  float inv = rsqrtf(var + 1e-5f);
  for (int j = tid; j < D_; j += 256)
    out[(size_t)row * D_ + j] = f2bf((xr[j] - mu) * inv * w[j] + b[j]);
}

// ---------------- repack qkv fp32 [BT,3D] -> q/k/v bf16 [B*H, T, DK] ----------------
__global__ void repack_qkv(const float* __restrict__ qkv, __bf16* __restrict__ q,
                           __bf16* __restrict__ k, __bf16* __restrict__ v) {
  int i = blockIdx.x * blockDim.x + threadIdx.x;
  if (i >= B_ * T_ * D_) return;
  int b   = i / (T_ * D_);
  int rem = i % (T_ * D_);
  int t   = rem / D_;
  int j   = rem % D_;
  int hh  = j / DK_;
  int d   = j % DK_;
  const float* src = qkv + (size_t)(b * T_ + t) * (3 * D_);
  size_t dst = ((size_t)(b * H_ + hh) * T_ + t) * DK_ + d;
  q[dst] = f2bf(src[j]);
  k[dst] = f2bf(src[D_ + j]);
  v[dst] = f2bf(src[2 * D_ + j]);
}

// ---------------- WMMA bf16 GEMM: out[M,N] = act[M,K] @ W[N,K]^T ----------------
// 256 threads = 8 waves; block tile 64x128; wave tile 32x32 = 2x2 wmma frags; BK=32.
// Double-buffered LDS, staged with global_load_async_to_lds_b128 (ASYNCcnt).
// A staged as bf16 directly; W staged as raw fp32 and converted at fragment load.
template <int ACT, int OUTBF, int RESID, int HASB>
__global__ __launch_bounds__(256)
void gemm_wmma(const __bf16* __restrict__ A, const float* __restrict__ W,
               const float* __restrict__ bias, const float* __restrict__ resid,
               void* __restrict__ outp, int M, int N, int K) {
  __shared__ alignas(16) __bf16 As[2][64][32];     // 2 x 4 KB
  __shared__ alignas(16) float  Bsf[2][128][32];   // 2 x 16 KB (raw W tile, [n][k])
  const int tid  = threadIdx.x;
  const int lane = tid & 31;
  const int wid  = tid >> 5;
  const int hlf  = lane >> 4, lr = lane & 15;
  const int wm   = wid >> 2, wn = wid & 3;      // wave grid 2x4
  const int m0   = blockIdx.y * 64;
  const int n0   = blockIdx.x * 128;

  const v8f zero = {0.f, 0.f, 0.f, 0.f, 0.f, 0.f, 0.f, 0.f};
  v8f acc[2][2];
  acc[0][0] = zero; acc[0][1] = zero; acc[1][0] = zero; acc[1][1] = zero;

  auto stage = [&](int kt, int buf) {
    // A tile: 64x32 bf16 = 4 KB, one b128 per thread
    {
      const int row = tid >> 2, col = (tid & 3) * 8;
      const __bf16* gp = A + (size_t)(m0 + row) * K + kt * 32 + col;
      async_b128(lds_addr(&As[buf][0][0]) + (unsigned)tid * 16u, gp);
    }
    // W tile: 128x32 fp32 = 16 KB, four b128 per thread; EXEC-masked tail + zero fill
#pragma unroll
    for (int s = 0; s < 4; ++s) {
      const int idx = tid + s * 256;            // 0..1023
      const int row = idx >> 3, c4 = (idx & 7) * 4;
      if (n0 + row < N) {
        const float* gp = W + (size_t)(n0 + row) * K + kt * 32 + c4;
        async_b128(lds_addr(&Bsf[buf][0][0]) + (unsigned)idx * 16u, gp);
      } else {
        float4 z = {0.f, 0.f, 0.f, 0.f};
        *(float4*)&Bsf[buf][row][c4] = z;
      }
    }
  };

  const int nk = K >> 5;
  stage(0, 0);
  for (int kt = 0; kt < nk; ++kt) {
    const int buf = kt & 1;
    wait_async0();
    __syncthreads();                 // buffer[buf] ready for all waves
    if (kt + 1 < nk) stage(kt + 1, buf ^ 1);   // overlap DMA with compute

    v16bf af[2], bfm[2];
#pragma unroll
    for (int i = 0; i < 2; ++i) {
      const int row = wm * 32 + i * 16 + lr;
#pragma unroll
      for (int e = 0; e < 16; ++e) af[i][e] = As[buf][row][apat(hlf, e)];
    }
#pragma unroll
    for (int j = 0; j < 2; ++j) {
      const int col = wn * 32 + j * 16 + lr;
#pragma unroll
      for (int e = 0; e < 16; ++e) bfm[j][e] = f2bf(Bsf[buf][col][hlf * 16 + e]);
    }
#pragma unroll
    for (int i = 0; i < 2; ++i)
#pragma unroll
      for (int j = 0; j < 2; ++j)
        acc[i][j] = __builtin_amdgcn_wmma_f32_16x16x32_bf16(
            false, af[i], false, bfm[j], (short)0, acc[i][j], false, false);
  }

  float*  of = (float*)outp;
  __bf16* ob = (__bf16*)outp;
#pragma unroll
  for (int i = 0; i < 2; ++i)
#pragma unroll
    for (int j = 0; j < 2; ++j)
#pragma unroll
      for (int r = 0; r < 8; ++r) {
        int rg = m0 + wm * 32 + i * 16 + hlf * 8 + r;
        int cg = n0 + wn * 32 + j * 16 + lr;
        if (cg < N) {
          float v = acc[i][j][r];
          if (HASB)  v += bias[cg];
          if (RESID) v += resid[(size_t)rg * N + cg];
          if (ACT)   v = 0.5f * v * (1.0f + erff(v * 0.70710678118654752f));
          if (OUTBF) ob[(size_t)rg * N + cg] = f2bf(v);
          else       of[(size_t)rg * N + cg] = v;
        }
      }
}

// ---------------- flash attention: 1 wave per (b,h, 16-query tile) ----------------
// __launch_bounds__(32,1): single wave per block, full VGPR budget -> no spills.
__global__ __launch_bounds__(32, 1)
void attn_wmma(const __bf16* __restrict__ q, const __bf16* __restrict__ k,
               const __bf16* __restrict__ vv, __bf16* __restrict__ y) {
  __shared__ alignas(16) __bf16 Ks[32][64];
  __shared__ alignas(16) __bf16 Vs[32][64];
  __shared__ alignas(16) __bf16 Ps[16][32];
  const int lane = threadIdx.x;
  const int hlf = lane >> 4, lr = lane & 15;
  const int bh = blockIdx.y;           // b*H + h
  const int q0 = blockIdx.x * 16;
  const int b = bh / H_, hh = bh % H_;
  const __bf16* qb = q  + (size_t)bh * T_ * DK_;
  const __bf16* kb = k  + (size_t)bh * T_ * DK_;
  const __bf16* vb = vv + (size_t)bh * T_ * DK_;

  // Q fragments: 16x64 = two 16x32 A frags
  v16bf aq[2];
#pragma unroll
  for (int kk = 0; kk < 2; ++kk)
#pragma unroll
    for (int e = 0; e < 16; ++e)
      aq[kk][e] = qb[(size_t)(q0 + lr) * DK_ + kk * 32 + apat(hlf, e)];

  const v8f zero = {0.f, 0.f, 0.f, 0.f, 0.f, 0.f, 0.f, 0.f};
  v8f o[4]; o[0] = zero; o[1] = zero; o[2] = zero; o[3] = zero;
  float mr[8], lw[8];
#pragma unroll
  for (int r = 0; r < 8; ++r) { mr[r] = -1e30f; lw[r] = 0.f; }

  const int nch = (q0 + 16 + 31) / 32;   // causal: keys up to q0+15
  for (int c = 0; c < nch; ++c) {
    const int c0 = c * 32;
    __syncthreads();
    // async-stage K/V chunk (32 rows x 64 bf16 each): 8 b128 per lane per tensor
    {
      const unsigned kbase = lds_addr(&Ks[0][0]) + (unsigned)lane * 128u;
      const unsigned vbase = lds_addr(&Vs[0][0]) + (unsigned)lane * 128u;
      const __bf16* kg = kb + (size_t)(c0 + lane) * DK_;
      const __bf16* vg = vb + (size_t)(c0 + lane) * DK_;
#pragma unroll
      for (int j = 0; j < 8; ++j) {
        async_b128(kbase + (unsigned)j * 16u, kg + j * 8);
        async_b128(vbase + (unsigned)j * 16u, vg + j * 8);
      }
    }
    wait_async0();
    __syncthreads();

    // S = Q (16x64) @ K^T (64x32): B[kc][n] = K[c0+n][kc]
    // One K-fragment live at a time to keep VGPR pressure low.
    v8f s01[2];
#pragma unroll
    for (int sub = 0; sub < 2; ++sub) {
      const int col = sub * 16 + lr;
      v16bf bk;
#pragma unroll
      for (int e = 0; e < 16; ++e) bk[e] = Ks[col][hlf * 16 + e];
      v8f t = __builtin_amdgcn_wmma_f32_16x16x32_bf16(
          false, aq[0], false, bk, (short)0, zero, false, false);
#pragma unroll
      for (int e = 0; e < 16; ++e) bk[e] = Ks[col][32 + hlf * 16 + e];
      s01[sub] = __builtin_amdgcn_wmma_f32_16x16x32_bf16(
          false, aq[1], false, bk, (short)0, t, false, false);
    }

    // scale + causal mask + online softmax (DPP reductions) + direct P store to LDS
#pragma unroll
    for (int r = 0; r < 8; ++r) {
      float v0 = s01[0][r] * 0.125f;           // 1/sqrt(64)
      float v1 = s01[1][r] * 0.125f;
      const int row = q0 + hlf * 8 + r;
      if (c0 + lr > row)      v0 = -1e30f;
      if (c0 + 16 + lr > row) v1 = -1e30f;
      const float mx = red16_max(fmaxf(v0, v1));
      const float mn = fmaxf(mr[r], mx);
      const float corr = __expf(mr[r] - mn);
      const float p0 = __expf(v0 - mn), p1 = __expf(v1 - mn);
      Ps[hlf * 8 + r][lr]      = f2bf(p0);     // C-layout -> LDS immediately
      Ps[hlf * 8 + r][16 + lr] = f2bf(p1);
      const float rs = red16_sum(p0 + p1);
      lw[r] = lw[r] * corr + rs;
      mr[r] = mn;
#pragma unroll
      for (int s2 = 0; s2 < 4; ++s2) o[s2][r] *= corr;
    }
    __syncthreads();

    // O += P @ V  (A-layout reload of P from LDS; one V fragment at a time)
    v16bf ap;
#pragma unroll
    for (int e = 0; e < 16; ++e) ap[e] = Ps[lr][apat(hlf, e)];
#pragma unroll
    for (int s2 = 0; s2 < 4; ++s2) {
      v16bf bv;
#pragma unroll
      for (int e = 0; e < 16; ++e) bv[e] = Vs[hlf * 16 + e][s2 * 16 + lr];
      o[s2] = __builtin_amdgcn_wmma_f32_16x16x32_bf16(
          false, ap, false, bv, (short)0, o[s2], false, false);
    }
  }

  // normalize + write y bf16 [B,T,D]
#pragma unroll
  for (int r = 0; r < 8; ++r) {
    const float inv = 1.0f / lw[r];
    const size_t rowo = (size_t)(b * T_ + q0 + hlf * 8 + r) * D_ + hh * DK_;
#pragma unroll
    for (int s2 = 0; s2 < 4; ++s2)
      y[rowo + s2 * 16 + lr] = f2bf(o[s2][r] * inv);
  }
}

// ---------------- driver ----------------
extern "C" void kernel_launch(void* const* d_in, const int* in_sizes, int n_in,
                              void* d_out, int out_size, void* d_ws, size_t ws_size,
                              hipStream_t stream) {
  (void)in_sizes; (void)n_in; (void)out_size; (void)ws_size;
  const int*   idx  = (const int*)d_in[0];
  const float* tok  = (const float*)d_in[1];
  const float* pos  = (const float*)d_in[2];
  const float* ln1w = (const float*)d_in[3];
  const float* ln1b = (const float*)d_in[4];
  const float* qkvw = (const float*)d_in[5];
  const float* qkvb = (const float*)d_in[6];
  const float* pjw  = (const float*)d_in[7];
  const float* pjb  = (const float*)d_in[8];
  const float* ln2w = (const float*)d_in[9];
  const float* ln2b = (const float*)d_in[10];
  const float* f1w  = (const float*)d_in[11];
  const float* f1b  = (const float*)d_in[12];
  const float* f2w  = (const float*)d_in[13];
  const float* f2b  = (const float*)d_in[14];
  const float* lnfw = (const float*)d_in[15];
  const float* lnfb = (const float*)d_in[16];
  const float* hdw  = (const float*)d_in[17];

  char* ws = (char*)d_ws;
  float*  x    = (float*)(ws + 0);                  // 2048*768 f32   = 6291456 B
  float*  qkv  = (float*)(ws + 6291456);            // 2048*2304 f32  = 18874368 B
  __bf16* h    = (__bf16*)(ws + 25165824);          // 2048*768 bf16  = 3145728 B
  __bf16* qb   = (__bf16*)(ws + 28311552);          // 3145728 B
  __bf16* kb   = (__bf16*)(ws + 31457280);          // 3145728 B
  __bf16* vb   = (__bf16*)(ws + 34603008);          // 3145728 B
  __bf16* y    = (__bf16*)(ws + 37748736);          // 3145728 B
  __bf16* ffa  = (__bf16*)(ws + 40894464);          // 2048*4096 bf16 = 16777216 B  (end 57671680)

  const int M = B_ * T_;                             // 2048
  const dim3 blk256(256), blk32(32);

  embed_k<<<M, 256, 0, stream>>>(idx, tok, pos, x);

  for (int l = 0; l < L_; ++l) {
    ln_bf16<<<M, 256, 0, stream>>>(x, ln1w + (size_t)l * D_, ln1b + (size_t)l * D_, h);
    gemm_wmma<0, 0, 0, 1><<<dim3(18, 32), blk256, 0, stream>>>(
        h, qkvw + (size_t)l * 3 * D_ * D_, qkvb + (size_t)l * 3 * D_, nullptr,
        (void*)qkv, M, 3 * D_, D_);
    repack_qkv<<<(B_ * T_ * D_ + 255) / 256, blk256, 0, stream>>>(qkv, qb, kb, vb);
    attn_wmma<<<dim3(T_ / 16, B_ * H_), blk32, 0, stream>>>(qb, kb, vb, y);
    gemm_wmma<0, 0, 1, 1><<<dim3(6, 32), blk256, 0, stream>>>(
        y, pjw + (size_t)l * D_ * D_, pjb + (size_t)l * D_, x, (void*)x, M, D_, D_);
    ln_bf16<<<M, 256, 0, stream>>>(x, ln2w + (size_t)l * D_, ln2b + (size_t)l * D_, h);
    gemm_wmma<1, 1, 0, 1><<<dim3(32, 32), blk256, 0, stream>>>(
        h, f1w + (size_t)l * DFF_ * D_, f1b + (size_t)l * DFF_, nullptr,
        (void*)ffa, M, DFF_, D_);
    gemm_wmma<0, 0, 1, 1><<<dim3(6, 32), blk256, 0, stream>>>(
        ffa, f2w + (size_t)l * D_ * DFF_, f2b + (size_t)l * D_, x, (void*)x, M, D_, DFF_);
  }

  ln_bf16<<<M, 256, 0, stream>>>(x, lnfw, lnfb, h);
  gemm_wmma<0, 0, 0, 0><<<dim3((V_ + 127) / 128, 32), blk256, 0, stream>>>(
      h, hdw, nullptr, nullptr, d_out, M, V_, D_);
}
